// PhysTTT_62989990363334
// MI455X (gfx1250) — matrix operand order repaired and checked by reading, not compiled, not run on hardware
//
#include <hip/hip_runtime.h>
#include <hip/hip_bf16.h>
#include <math.h>

typedef __attribute__((ext_vector_type(16))) __bf16 bf16x16;
typedef __attribute__((ext_vector_type(8)))  float  f32x8;

// ---------------------------------------------------------------- helpers
__device__ __forceinline__ float gelu_f(float x) {
    float x3 = x * x * x;
    return 0.5f * x * (1.0f + tanhf(0.7978845608f * (x + 0.044715f * x3)));
}
__device__ __forceinline__ float gelu_bwd_f(float x) {
    float x2 = x * x;
    float t = tanhf(0.79788456f * x * (1.0f + 0.044715f * x2));
    return 0.5f * x * ((1.0f - t * t) * (0.79788456f + 0.1070322243f * x2)) + 0.5f * (1.0f + t);
}
__device__ __forceinline__ float sigm_f(float x) { return 1.0f / (1.0f + __expf(-x)); }

// A-fragment loader: 16x32 bf16 A tile slice for one lane.
// Per ISA 7.12.2: lane l (row = l%16), elem i -> K = kk + i + 8*(i>=8) + 8*(l>=16)
// => two contiguous 8-float runs at kk+8*hl and kk+16+8*hl.
__device__ __forceinline__ bf16x16 load_a_frag(const float* __restrict__ Arow) {
    const float4* p = (const float4*)Arow;
    float4 a0 = p[0], a1 = p[1], a2 = p[4], a3 = p[5];
    bf16x16 r;
    r[0] = (__bf16)a0.x; r[1] = (__bf16)a0.y; r[2]  = (__bf16)a0.z; r[3]  = (__bf16)a0.w;
    r[4] = (__bf16)a1.x; r[5] = (__bf16)a1.y; r[6]  = (__bf16)a1.z; r[7]  = (__bf16)a1.w;
    r[8] = (__bf16)a2.x; r[9] = (__bf16)a2.y; r[10] = (__bf16)a2.z; r[11] = (__bf16)a2.w;
    r[12] = (__bf16)a3.x; r[13] = (__bf16)a3.y; r[14] = (__bf16)a3.z; r[15] = (__bf16)a3.w;
    return r;
}
// B-fragment from LDS tile Bs[32][65]: lane l (col = sub*16 + l%16), elem i -> K = i + 8*(i>=8) + 8*(l>=16)
__device__ __forceinline__ bf16x16 load_b_frag(const float* Bs, int hl, int lr, int sub) {
    bf16x16 r;
#pragma unroll
    for (int i = 0; i < 16; ++i) r[i] = (__bf16)Bs[(i + 8 * hl + 8 * (i >= 8)) * 65 + sub * 16 + lr];
    return r;
}

// ---------------------------------------------------------------- elementwise utils
__global__ __launch_bounds__(256) void fill0_k(float* p, int n) {
    int i = blockIdx.x * 256 + threadIdx.x;
    if (i < n) p[i] = 0.0f;
}
__global__ __launch_bounds__(256) void avg2_k(const float* a, const float* b, float* o, int n) {
    int i = blockIdx.x * 256 + threadIdx.x;
    if (i < n) o[i] = 0.5f * (a[i] + b[i]);
}

// ---------------------------------------------------------------- stem: conv7x7 + BN + ReLU + maxpool2
__global__ __launch_bounds__(256) void conv7_bn_relu_pool(
    const float* __restrict__ in, const float* __restrict__ w, const float* __restrict__ cb,
    const float* __restrict__ bn, float* __restrict__ out,
    int N, int IC, int OC, int IH, int stride, int PH)
{
    long idx = (long)blockIdx.x * 256 + threadIdx.x;
    long total = (long)N * OC * PH * PH;
    if (idx >= total) return;
    int px = idx % PH, py = (idx / PH) % PH;
    int oc = (idx / ((long)PH * PH)) % OC;
    int n  = idx / ((long)PH * PH * OC);
    float gm = bn[oc], be = bn[OC + oc], mn = bn[2 * OC + oc], vr = bn[3 * OC + oc];
    float sc = gm * rsqrtf(vr + 1e-5f);
    float best = -3.4e38f;
    for (int dy = 0; dy < 2; ++dy)
    for (int dx = 0; dx < 2; ++dx) {
        int oy = py * 2 + dy, ox = px * 2 + dx;
        float a = cb[oc];
        for (int ic = 0; ic < IC; ++ic)
        for (int ky = 0; ky < 7; ++ky) {
            int iy = oy * stride + ky - 3;
            if (iy < 0 || iy >= IH) continue;
            for (int kx = 0; kx < 7; ++kx) {
                int ix = ox * stride + kx - 3;
                if (ix < 0 || ix >= IH) continue;
                a = fmaf(in[(((long)n * IC + ic) * IH + iy) * IH + ix],
                         w[((oc * IC + ic) * 7 + ky) * 7 + kx], a);
            }
        }
        float r = fmaxf((a - mn) * sc + be, 0.0f);
        best = fmaxf(best, r);
    }
    out[(((long)n * OC + oc) * PH + py) * PH + px] = best;
}

// frame-difference value computed on the fly (Fusion_Stem diffs, 12 channels)
__device__ __forceinline__ float xval_f(const float* x, int b, int dd, int c, int y, int xx) {
    return x[((((long)b * 256 + dd) * 3 + c) * 96 + y) * 96 + xx];
}
__device__ __forceinline__ float diffval_f(const float* x, int b, int d, int ch, int y, int xx) {
    int grp = ch / 3, c = ch % 3;
    int da, db;
    if (grp == 0)      { da = (d >= 1) ? d - 1 : 0;     db = (d >= 2) ? d - 2 : 0; }
    else if (grp == 1) { da = d;                        db = (d >= 1) ? d - 1 : 0; }
    else if (grp == 2) { da = (d <= 254) ? d + 1 : 255; db = d; }
    else               { da = (d <= 253) ? d + 2 : 255; db = (d <= 254) ? d + 1 : 255; }
    return xval_f(x, b, da, c, y, xx) - xval_f(x, b, db, c, y, xx);
}

__global__ __launch_bounds__(256) void conv7_diff_pool(
    const float* __restrict__ x, const float* __restrict__ w, const float* __restrict__ cb,
    const float* __restrict__ bn, float* __restrict__ out)
{
    const int OC = 8, PH = 24;
    long idx = (long)blockIdx.x * 256 + threadIdx.x;
    long total = 512L * OC * PH * PH;
    if (idx >= total) return;
    int px = idx % PH, py = (idx / PH) % PH;
    int oc = (idx / (PH * PH)) % OC;
    int n  = idx / (PH * PH * OC);
    int b = n >> 8, d = n & 255;
    float gm = bn[oc], be = bn[OC + oc], mn = bn[2 * OC + oc], vr = bn[3 * OC + oc];
    float sc = gm * rsqrtf(vr + 1e-5f);
    float best = -3.4e38f;
    for (int dy = 0; dy < 2; ++dy)
    for (int dx = 0; dx < 2; ++dx) {
        int oy = py * 2 + dy, ox = px * 2 + dx;
        float a = cb[oc];
        for (int ic = 0; ic < 12; ++ic)
        for (int ky = 0; ky < 7; ++ky) {
            int iy = oy * 2 + ky - 3;
            if (iy < 0 || iy >= 96) continue;
            for (int kx = 0; kx < 7; ++kx) {
                int ix = ox * 2 + kx - 3;
                if (ix < 0 || ix >= 96) continue;
                a = fmaf(diffval_f(x, b, d, ic, iy, ix), w[((oc * 12 + ic) * 7 + ky) * 7 + kx], a);
            }
        }
        float r = fmaxf((a - mn) * sc + be, 0.0f);
        best = fmaxf(best, r);
    }
    out[(((long)n * OC + oc) * PH + py) * PH + px] = best;
}

__global__ __launch_bounds__(256) void conv3_bn(
    const float* __restrict__ in, const float* __restrict__ w, const float* __restrict__ cb,
    const float* __restrict__ bn, float* __restrict__ out)
{
    long idx = (long)blockIdx.x * 256 + threadIdx.x;
    long total = 512L * 64 * 144;
    if (idx >= total) return;
    int px = idx % 12, py = (idx / 12) % 12;
    int oc = (idx / 144) % 64;
    int n  = idx / (144 * 64);
    float a = cb[oc];
    for (int ic = 0; ic < 16; ++ic)
    for (int ky = 0; ky < 3; ++ky) {
        int iy = py + ky - 1;
        if (iy < 0 || iy >= 12) continue;
        for (int kx = 0; kx < 3; ++kx) {
            int ix = px + kx - 1;
            if (ix < 0 || ix >= 12) continue;
            a = fmaf(in[(((long)n * 16 + ic) * 12 + iy) * 12 + ix],
                     w[((oc * 16 + ic) * 3 + ky) * 3 + kx], a);
        }
    }
    float gm = bn[oc], be = bn[64 + oc], mn = bn[128 + oc], vr = bn[192 + oc];
    out[((long)n * 64 + oc) * 144 + py * 12 + px] = (a - mn) * gm * rsqrtf(vr + 1e-5f) + be;
}

// attention mask + spatial mean -> sig, stored channel-major [B][64][256]
__global__ __launch_bounds__(64) void attn_sig(const float* __restrict__ feat, float* __restrict__ sigT) {
    int n = blockIdx.x, c = threadIdx.x;
    const float* f = feat + ((long)n * 64 + c) * 144;
    float s1 = 0.0f, s2 = 0.0f;
    for (int i = 0; i < 144; ++i) { float z = f[i]; float s = sigm_f(z); s1 += s; s2 = fmaf(z, s, s2); }
    int b = n >> 8, d = n & 255;
    sigT[(long)b * 16384 + c * 256 + d] = 0.5f * s2 / s1;
}

// ---------------------------------------------------------------- bf16 WMMA GEMM, f32 accumulate
// CONTRACT: M%64==0, N%64==0, K%32==0, lda/ldb/ldc%4==0, 16B-aligned pointers.
// Block = 4 waves = 64x64 C tile; wave w owns rows [bm+16w, +16), all 64 cols.
// B tile (32x64) staged in LDS (stride 65: conflict-free), shared by the 4 waves.
__global__ __launch_bounds__(128) void gemm_bf16_wmma(
    const float* __restrict__ A, long sA, int lda,
    const float* __restrict__ B, long sB, int ldb,
    float* __restrict__ C, long sC, int ldc,
    int M, int N, int K,
    const float* __restrict__ biasM, const float* __restrict__ biasN, int relu)
{
    __shared__ float Bs[32 * 65];
    A += (long)blockIdx.y * sA; B += (long)blockIdx.y * sB; C += (long)blockIdx.y * sC;
    int wv = threadIdx.x >> 5, lane = threadIdx.x & 31;
    int hl = lane >> 4, lr = lane & 15;
    int nbn = N >> 6;
    int bm = (blockIdx.x / nbn) * 64;
    int bn = (blockIdx.x % nbn) * 64;
    int arow = bm + wv * 16 + lr;
    f32x8 acc0 = {0.f,0.f,0.f,0.f,0.f,0.f,0.f,0.f};
    f32x8 acc1 = acc0, acc2 = acc0, acc3 = acc0;
    for (int kk = 0; kk < K; kk += 32) {
#pragma unroll
        for (int j = threadIdx.x; j < 2048; j += 128) {
            int kr = j >> 6, cl = j & 63;
            Bs[kr * 65 + cl] = B[(long)(kk + kr) * ldb + bn + cl];
        }
        __syncthreads();
        bf16x16 af = load_a_frag(A + (long)arow * lda + kk + hl * 8);
        bf16x16 b0 = load_b_frag(Bs, hl, lr, 0);
        bf16x16 b1 = load_b_frag(Bs, hl, lr, 1);
        bf16x16 b2 = load_b_frag(Bs, hl, lr, 2);
        bf16x16 b3 = load_b_frag(Bs, hl, lr, 3);
        acc0 = __builtin_amdgcn_wmma_f32_16x16x32_bf16(false, af, false, b0, (short)0, acc0, false, false);
        acc1 = __builtin_amdgcn_wmma_f32_16x16x32_bf16(false, af, false, b1, (short)0, acc1, false, false);
        acc2 = __builtin_amdgcn_wmma_f32_16x16x32_bf16(false, af, false, b2, (short)0, acc2, false, false);
        acc3 = __builtin_amdgcn_wmma_f32_16x16x32_bf16(false, af, false, b3, (short)0, acc3, false, false);
        __syncthreads();
    }
#pragma unroll
    for (int s = 0; s < 4; ++s) {
        f32x8 acc = (s == 0) ? acc0 : (s == 1) ? acc1 : (s == 2) ? acc2 : acc3;
        int gn = bn + s * 16 + lr;
#pragma unroll
        for (int r = 0; r < 8; ++r) {
            int gm = bm + wv * 16 + r + hl * 8;
            float v = acc[r];
            if (biasM) v += biasM[gm];
            if (biasN) v += biasN[gn];
            if (relu) v = fmaxf(v, 0.0f);
            C[(long)gm * ldc + gn] = v;
        }
    }
}

// conv1d (k=3, pad=1, 256ch) as implicit-im2col WMMA GEMM: M=oc256, N=t256, K=768.
// Boundary (t=-1 / t=256) handled branchlessly at LDS staging time.
__global__ __launch_bounds__(128) void conv1d_k3_wmma(
    const float* __restrict__ W, const float* __restrict__ bias,
    const float* __restrict__ Hin, float* __restrict__ Hout,
    float* __restrict__ Acc, int addAcc)
{
    __shared__ float Bs[32 * 65];
    const float* Hb = Hin + (long)blockIdx.y * 65536;
    float* Ho = Hout + (long)blockIdx.y * 65536;
    float* Ac = Acc + (long)blockIdx.y * 65536;
    int wv = threadIdx.x >> 5, lane = threadIdx.x & 31;
    int hl = lane >> 4, lr = lane & 15;
    int bm = (blockIdx.x >> 2) * 64;
    int bn = (blockIdx.x & 3) * 64;
    int arow = bm + wv * 16 + lr;
    f32x8 acc0 = {0.f,0.f,0.f,0.f,0.f,0.f,0.f,0.f};
    f32x8 acc1 = acc0, acc2 = acc0, acc3 = acc0;
    for (int kk = 0; kk < 768; kk += 32) {
#pragma unroll
        for (int j = threadIdx.x; j < 2048; j += 128) {
            int kr = j >> 6, cl = j & 63;
            int kb = kk + kr;
            int ic = kb / 3, dk = kb - ic * 3;
            int tt = bn + cl + dk - 1;
            int tc = min(max(tt, 0), 255);
            float v = Hb[ic * 256 + tc];
            Bs[kr * 65 + cl] = (tt == tc) ? v : 0.0f;
        }
        __syncthreads();
        bf16x16 af = load_a_frag(W + (long)arow * 768 + kk + hl * 8);
        bf16x16 b0 = load_b_frag(Bs, hl, lr, 0);
        bf16x16 b1 = load_b_frag(Bs, hl, lr, 1);
        bf16x16 b2 = load_b_frag(Bs, hl, lr, 2);
        bf16x16 b3 = load_b_frag(Bs, hl, lr, 3);
        acc0 = __builtin_amdgcn_wmma_f32_16x16x32_bf16(false, af, false, b0, (short)0, acc0, false, false);
        acc1 = __builtin_amdgcn_wmma_f32_16x16x32_bf16(false, af, false, b1, (short)0, acc1, false, false);
        acc2 = __builtin_amdgcn_wmma_f32_16x16x32_bf16(false, af, false, b2, (short)0, acc2, false, false);
        acc3 = __builtin_amdgcn_wmma_f32_16x16x32_bf16(false, af, false, b3, (short)0, acc3, false, false);
        __syncthreads();
    }
#pragma unroll
    for (int s = 0; s < 4; ++s) {
        f32x8 acc = (s == 0) ? acc0 : (s == 1) ? acc1 : (s == 2) ? acc2 : acc3;
        int gn = bn + s * 16 + lr;
#pragma unroll
        for (int r = 0; r < 8; ++r) {
            int gm = bm + wv * 16 + r + hl * 8;
            float v = fmaxf(acc[r] + bias[gm], 0.0f);
            Ho[(long)gm * 256 + gn] = v;
            if (addAcc) Ac[(long)gm * 256 + gn] += v;
        }
    }
}

// ---------------------------------------------------------------- layernorm over 256 (mode1 = transposed [b][c][l] input)
__global__ __launch_bounds__(256) void layernorm256(
    const float* __restrict__ in, const float* __restrict__ g, const float* __restrict__ b,
    float* __restrict__ out, int mode)
{
    __shared__ float red[256];
    int row = blockIdx.x, t = threadIdx.x;
    long off = mode ? ((long)(row >> 8) * 65536 + (row & 255) + (long)t * 256)
                    : ((long)row * 256 + t);
    float x = in[off];
    red[t] = x; __syncthreads();
    for (int s = 128; s > 0; s >>= 1) { if (t < s) red[t] += red[t + s]; __syncthreads(); }
    float mu = red[0] * (1.0f / 256.0f); __syncthreads();
    float dx = x - mu;
    red[t] = dx * dx; __syncthreads();
    for (int s = 128; s > 0; s >>= 1) { if (t < s) red[t] += red[t + s]; __syncthreads(); }
    float var = red[0] * (1.0f / 256.0f);
    out[(long)row * 256 + t] = dx * rsqrtf(var + 1e-5f) * g[t] + b[t];
}

// eta_lr = BASE_LR * sigmoid(hidden . lr_w + lr_b) / HD, layout [b][h][l]
__global__ __launch_bounds__(256) void eta_lr_kernel(
    const float* __restrict__ hidden, const float* __restrict__ lrw,
    const float* __restrict__ lrb, float* __restrict__ elr)
{
    int idx = blockIdx.x * 256 + threadIdx.x;
    if (idx >= 2048) return;
    int b = idx >> 10, h = (idx >> 8) & 3, l = idx & 255;
    const float* hr = hidden + ((long)b * 256 + l) * 256;
    const float* w = lrw + h * 256;
    float s = lrb[h];
    for (int d = 0; d < 256; ++d) s = fmaf(hr[d], w[d], s);
    elr[idx] = 0.1f * sigm_f(s) / 64.0f;
}

// ---------------------------------------------------------------- TTT-MLP sequential scan: 1 workgroup per (b,head)
__global__ __launch_bounds__(256) void ttt_scan(
    const float* __restrict__ Qm, const float* __restrict__ Km, const float* __restrict__ Vm,
    const float* __restrict__ elr, const float* __restrict__ tib,
    const float* __restrict__ lnw, const float* __restrict__ lnb,
    const float* __restrict__ W1i, const float* __restrict__ B1i,
    const float* __restrict__ W2i, const float* __restrict__ B2i,
    float* __restrict__ scanW, float* __restrict__ O)
{
    const int bh = blockIdx.x, b = bh >> 2, h = bh & 3;
    const int t = threadIdx.x;
    float* W1 = scanW + (long)bh * (64 * 256 + 256 * 64);
    float* W2 = W1 + 64 * 256;
    __shared__ float b1[256], b2[64], g[64], bb[64];
    __shared__ float q[4][64], k[4][64], v[4][64];
    __shared__ float Z1[4][256], X2[4][256], gZ1[4][256], Z1b[4][256], X2b[4][256];
    __shared__ float Z2[4][64], gZ2[4][64], Z2b[4][64];
    __shared__ float A1[4][4], A2[4][4], et[4][4], le[4], tok[4], ee[4];

    for (int i = t; i < 64 * 256; i += 256) W1[i] = W1i[(long)h * 64 * 256 + i];
    for (int i = t; i < 256 * 64; i += 256) W2[i] = W2i[(long)h * 256 * 64 + i];
    b1[t] = B1i[h * 256 + t];
    if (t < 64) { b2[t] = B2i[h * 64 + t]; g[t] = lnw[h * 64 + t]; bb[t] = lnb[h * 64 + t]; }
    if (t < 4) tok[t] = fmaxf(1.0f / (float)(t + 1) + tib[t], 0.0f);
    __syncthreads();

    for (int c = 0; c < 64; ++c) {
        { // S0: load q,k,v chunk, eta row
            int m = t >> 6, d = t & 63;
            long r = (((long)b * 256) + c * 4 + m) * 256 + h * 64 + d;
            q[m][d] = Qm[r]; k[m][d] = Km[r]; v[m][d] = Vm[r];
            if (t < 4) ee[t] = elr[bh * 256 + c * 4 + t];
        }
        __syncthreads();
        if (t < 16) { // eta masked + last row
            int i_ = t >> 2, j_ = t & 3;
            et[i_][j_] = (j_ <= i_) ? tok[i_] * ee[j_] : 0.0f;
            if (i_ == 0) le[j_] = tok[3] * ee[j_];
        }
        { // S1: Z1 = k@W1 + b1 ; X2 = gelu(Z1)
            float za[4] = {0.f, 0.f, 0.f, 0.f};
            for (int d = 0; d < 64; ++d) {
                float w = W1[d * 256 + t];
#pragma unroll
                for (int m = 0; m < 4; ++m) za[m] = fmaf(k[m][d], w, za[m]);
            }
            float bj = b1[t];
#pragma unroll
            for (int m = 0; m < 4; ++m) { float z = za[m] + bj; Z1[m][t] = z; X2[m][t] = gelu_f(z); }
        }
        __syncthreads();
        { // S2: Z2 = X2@W2 + b2
            int m = t >> 6, j2 = t & 63;
            float a = b2[j2];
            for (int j = 0; j < 256; ++j) a = fmaf(X2[m][j], W2[j * 64 + j2], a);
            Z2[m][j2] = a;
        }
        __syncthreads();
        // S3: gZ2 = ln_l2_bwd(Z2, v-k) on threads 0..3 ; A1 = tril(q k^T) on threads 16..31
        if (t < 4) {
            float mu = 0.f, sq = 0.f;
            for (int j = 0; j < 64; ++j) { float z = Z2[t][j]; mu += z; sq = fmaf(z, z, sq); }
            mu *= (1.0f / 64.0f);
            float stdv = sqrtf(sq * (1.0f / 64.0f) - mu * mu + 1e-6f);
            float inv = 1.0f / stdv;
            float sg = 0.f, sgx = 0.f;
            for (int j = 0; j < 64; ++j) {
                float xh = (Z2[t][j] - mu) * inv;
                float gxh = (g[j] * xh + bb[j] - (v[t][j] - k[t][j])) * g[j];
                gZ2[t][j] = gxh; sg += gxh; sgx = fmaf(gxh, xh, sgx);
            }
            float sc = 1.0f / (64.0f * stdv);
            for (int j = 0; j < 64; ++j) {
                float xh = (Z2[t][j] - mu) * inv;
                gZ2[t][j] = (64.0f * gZ2[t][j] - sg - xh * sgx) * sc;
            }
        } else if (t >= 16 && t < 32) {
            int i_ = (t - 16) >> 2, j_ = t & 3;
            float a = 0.f;
            if (j_ <= i_) for (int d = 0; d < 64; ++d) a = fmaf(q[i_][d], k[j_][d], a);
            A1[i_][j_] = a;
        }
        __syncthreads();
        { // S4: gZ1 = (gZ2 @ W2^T) * gelu_bwd(Z1)
            float za[4] = {0.f, 0.f, 0.f, 0.f};
            for (int j2 = 0; j2 < 64; ++j2) {
                float w = W2[t * 64 + j2];
#pragma unroll
                for (int m = 0; m < 4; ++m) za[m] = fmaf(gZ2[m][j2], w, za[m]);
            }
#pragma unroll
            for (int m = 0; m < 4; ++m) gZ1[m][t] = za[m] * gelu_bwd_f(Z1[m][t]);
        }
        __syncthreads();
        { // S5: Z1b = q@W1 + b1 - sum_i et*(A1+1)*gZ1 ; X2b = gelu
            float za[4] = {0.f, 0.f, 0.f, 0.f};
            for (int d = 0; d < 64; ++d) {
                float w = W1[d * 256 + t];
#pragma unroll
                for (int m = 0; m < 4; ++m) za[m] = fmaf(q[m][d], w, za[m]);
            }
            float bj = b1[t];
#pragma unroll
            for (int m = 0; m < 4; ++m) {
                float z = za[m] + bj;
#pragma unroll
                for (int i = 0; i < 4; ++i) z -= et[m][i] * (A1[m][i] + 1.0f) * gZ1[i][t];
                Z1b[m][t] = z; X2b[m][t] = gelu_f(z);
            }
        }
        __syncthreads();
        if (t < 16) { // S6: A2 = tril(X2b X2^T)
            int i_ = t >> 2, j_ = t & 3;
            float a = 0.f;
            if (j_ <= i_) for (int j = 0; j < 256; ++j) a = fmaf(X2b[i_][j], X2[j_][j], a);
            A2[i_][j_] = a;
        }
        __syncthreads();
        { // S7: Z2b
            int m = t >> 6, j2 = t & 63;
            float a = b2[j2];
            for (int j = 0; j < 256; ++j) a = fmaf(X2b[m][j], W2[j * 64 + j2], a);
#pragma unroll
            for (int i = 0; i < 4; ++i) a -= et[m][i] * (A2[m][i] + 1.0f) * gZ2[i][j2];
            Z2b[m][j2] = a;
        }
        __syncthreads();
        // S8: output row (t<4) + fast-weight updates (all threads)
        if (t < 4) {
            float mu = 0.f, sq = 0.f;
            for (int j = 0; j < 64; ++j) { float z = Z2b[t][j]; mu += z; sq = fmaf(z, z, sq); }
            mu *= (1.0f / 64.0f);
            float inv = 1.0f / sqrtf(sq * (1.0f / 64.0f) - mu * mu + 1e-6f);
            long rbase = (((long)b * 256) + c * 4 + t) * 256 + h * 64;
            for (int j = 0; j < 64; ++j) {
                float xh = (Z2b[t][j] - mu) * inv;
                O[rbase + j] = q[t][j] + g[j] * xh + bb[j];
            }
        }
        {
            float g0 = le[0] * gZ1[0][t], g1 = le[1] * gZ1[1][t];
            float g2 = le[2] * gZ1[2][t], g3 = le[3] * gZ1[3][t];
            for (int d = 0; d < 64; ++d)
                W1[d * 256 + t] -= g0 * k[0][d] + g1 * k[1][d] + g2 * k[2][d] + g3 * k[3][d];
            b1[t] -= (g0 + g1 + g2 + g3);
            int j2 = t & 63, jb = t >> 6;
            float c0 = le[0] * gZ2[0][j2], c1 = le[1] * gZ2[1][j2];
            float c2 = le[2] * gZ2[2][j2], c3 = le[3] * gZ2[3][j2];
            for (int jj = 0; jj < 64; ++jj) {
                int j = jb * 64 + jj;
                W2[j * 64 + j2] -= c0 * X2b[0][j] + c1 * X2b[1][j] + c2 * X2b[2][j] + c3 * X2b[3][j];
            }
            if (t < 64)
                b2[t] -= le[0] * gZ2[0][t] + le[1] * gZ2[1][t] + le[2] * gZ2[2][t] + le[3] * gZ2[3][t];
        }
        __syncthreads();
    }
}

// final FC3: [512,64] @ [64,1] + b -> d_out [B,D]
__global__ __launch_bounds__(256) void fc3_kernel(
    const float* __restrict__ in, const float* __restrict__ w, const float* __restrict__ b,
    float* __restrict__ out)
{
    int r = blockIdx.x * 256 + threadIdx.x;
    if (r >= 512) return;
    const float* row = in + (long)r * 64;
    float a = b[0];
    for (int i = 0; i < 64; ++i) a = fmaf(row[i], w[i], a);
    out[r] = a;
}

// ---------------------------------------------------------------- host
extern "C" void kernel_launch(void* const* d_in, const int* in_sizes, int n_in,
                              void* d_out, int out_size, void* d_ws, size_t ws_size,
                              hipStream_t stream) {
    const float* x      = (const float*)d_in[0];
    const float* s11_w  = (const float*)d_in[1];  const float* s11_b = (const float*)d_in[2];
    const float* bn11   = (const float*)d_in[3];
    const float* s12_w  = (const float*)d_in[4];  const float* s12_b = (const float*)d_in[5];
    const float* bn12   = (const float*)d_in[6];
    const float* s21_w  = (const float*)d_in[7];  const float* s21_b = (const float*)d_in[8];
    const float* bn21   = (const float*)d_in[9];
    const float* s22_w  = (const float*)d_in[10]; const float* s22_b = (const float*)d_in[11];
    const float* bn22   = (const float*)d_in[12];
    const float* s3_w   = (const float*)d_in[13]; const float* s3_b  = (const float*)d_in[14];
    const float* bn3    = (const float*)d_in[15];
    const float* c1_w   = (const float*)d_in[16]; const float* c1_b  = (const float*)d_in[17];
    const float* cb_w   = (const float*)d_in[18]; const float* cb_b  = (const float*)d_in[19];
    const float* ln_g   = (const float*)d_in[20]; const float* ln_b  = (const float*)d_in[21];
    const float* wq     = (const float*)d_in[22]; const float* wk    = (const float*)d_in[23];
    const float* wv     = (const float*)d_in[24]; const float* wo    = (const float*)d_in[25];
    const float* lr_w   = (const float*)d_in[26]; const float* lr_b  = (const float*)d_in[27];
    const float* tib    = (const float*)d_in[28];
    const float* tlnw   = (const float*)d_in[29]; const float* tlnb  = (const float*)d_in[30];
    const float* W1     = (const float*)d_in[31]; const float* B1    = (const float*)d_in[32];
    const float* W2     = (const float*)d_in[33]; const float* B2    = (const float*)d_in[34];
    const float* post_g = (const float*)d_in[35]; const float* post_b= (const float*)d_in[36];
    const float* fc_w   = (const float*)d_in[37]; const float* fc_b  = (const float*)d_in[38];
    const float* fc2_w  = (const float*)d_in[39]; const float* fc2_b = (const float*)d_in[40];
    const float* fc3_w  = (const float*)d_in[41]; const float* fc3_b = (const float*)d_in[42];

    float* ws = (float*)d_ws;
    size_t o = 0;
    auto alloc = [&](size_t n) { float* p = ws + o; o += n; return p; };
    float* xs   = alloc(512L * 8 * 24 * 24);
    float* xd   = alloc(512L * 8 * 24 * 24);
    float* pbuf = alloc(512L * 8 * 24 * 24);
    float* p1   = alloc(512L * 16 * 12 * 12);
    float* p2   = alloc(512L * 16 * 12 * 12);
    float* q3   = alloc(512L * 16 * 12 * 12);
    float* feat = alloc(512L * 64 * 144);
    float* sigT = alloc(2L * 64 * 256);
    float* hA   = alloc(2L * 256 * 256);
    float* hB   = alloc(2L * 256 * 256);
    float* accT = alloc(2L * 256 * 256);
    float* hid  = alloc(512L * 256);
    float* Qm   = alloc(512L * 256);
    float* Km   = alloc(512L * 256);
    float* Vm   = alloc(512L * 256);
    float* elr  = alloc(2048);
    float* scW  = alloc(8L * (64 * 256 + 256 * 64));
    float* Obuf = alloc(512L * 256);
    float* pln  = alloc(512L * 256);
    float* h2   = alloc(512L * 256);
    float* h3   = alloc(512L * 256);
    float* f1   = alloc(512L * 128);
    float* f2   = alloc(512L * 64);

    // stem
    {
        long tot = 512L * 8 * 24 * 24;
        conv7_bn_relu_pool<<<(tot + 255) / 256, 256, 0, stream>>>(x, s11_w, s11_b, bn11, xs, 512, 3, 8, 96, 2, 24);
        conv7_diff_pool<<<(tot + 255) / 256, 256, 0, stream>>>(x, s12_w, s12_b, bn12, xd);
        avg2_k<<<((int)tot + 255) / 256, 256, 0, stream>>>(xs, xd, pbuf, (int)tot);
    }
    {
        long tot = 512L * 16 * 12 * 12;
        conv7_bn_relu_pool<<<(tot + 255) / 256, 256, 0, stream>>>(pbuf, s21_w, s21_b, bn21, p1, 512, 8, 16, 24, 1, 12);
        conv7_bn_relu_pool<<<(tot + 255) / 256, 256, 0, stream>>>(xd,   s22_w, s22_b, bn22, p2, 512, 8, 16, 24, 1, 12);
        avg2_k<<<((int)tot + 255) / 256, 256, 0, stream>>>(p1, p2, q3, (int)tot);
    }
    conv3_bn<<<(512L * 64 * 144 + 255) / 256, 256, 0, stream>>>(q3, s3_w, s3_b, bn3, feat);
    attn_sig<<<512, 64, 0, stream>>>(feat, sigT);

    // ResNetTTT trunk (WMMA GEMMs): tiles 64x64 per block
    gemm_bf16_wmma<<<dim3(16, 2), 128, 0, stream>>>(c1_w, 0, 64, sigT, 64L * 256, 256,
                                                    hA, 256L * 256, 256, 256, 256, 64, c1_b, nullptr, 1);
    fill0_k<<<(131072 + 255) / 256, 256, 0, stream>>>(accT, 131072);
    float* cur = hA; float* oth = hB;
    for (int i = 0; i < 8; ++i) {
        conv1d_k3_wmma<<<dim3(16, 2), 128, 0, stream>>>(cb_w + (long)i * 256 * 768, cb_b + i * 256,
                                                        cur, oth, accT, (i & 1));
        float* tmp = cur; cur = oth; oth = tmp;
    }
    layernorm256<<<512, 256, 0, stream>>>(accT, ln_g, ln_b, hid, 1);

    // TTT-MLP
    eta_lr_kernel<<<8, 256, 0, stream>>>(hid, lr_w, lr_b, elr);
    gemm_bf16_wmma<<<dim3(32, 1), 128, 0, stream>>>(hid, 0, 256, wq, 0, 256, Qm, 0, 256, 512, 256, 256, nullptr, nullptr, 0);
    gemm_bf16_wmma<<<dim3(32, 1), 128, 0, stream>>>(hid, 0, 256, wk, 0, 256, Km, 0, 256, 512, 256, 256, nullptr, nullptr, 0);
    gemm_bf16_wmma<<<dim3(32, 1), 128, 0, stream>>>(hid, 0, 256, wv, 0, 256, Vm, 0, 256, 512, 256, 256, nullptr, nullptr, 0);
    ttt_scan<<<8, 256, 0, stream>>>(Qm, Km, Vm, elr, tib, tlnw, tlnb, W1, B1, W2, B2, scW, Obuf);
    layernorm256<<<512, 256, 0, stream>>>(Obuf, post_g, post_b, pln, 0);
    gemm_bf16_wmma<<<dim3(32, 1), 128, 0, stream>>>(pln, 0, 256, wo, 0, 256, h2, 0, 256, 512, 256, 256, nullptr, nullptr, 0);
    layernorm256<<<512, 256, 0, stream>>>(h2, ln_g, ln_b, h3, 0);

    // head
    gemm_bf16_wmma<<<dim3(16, 1), 128, 0, stream>>>(h3, 0, 256, fc_w, 0, 128, f1, 0, 128, 512, 128, 256, nullptr, fc_b, 0);
    gemm_bf16_wmma<<<dim3(8, 1), 128, 0, stream>>>(f1, 0, 128, fc2_w, 0, 64, f2, 0, 64, 512, 64, 128, nullptr, fc2_b, 0);
    fc3_kernel<<<2, 256, 0, stream>>>(f2, fc3_w, fc3_b, (float*)d_out);
}
